// GAT_36696200577053
// MI455X (gfx1250) — matrix operand order repaired and checked by reading, not compiled, not run on hardware
//
#include <hip/hip_runtime.h>
#include <hip/hip_bf16.h>

typedef __attribute__((ext_vector_type(16))) _Float16 v16h;
typedef __attribute__((ext_vector_type(8)))  float    v8f;

#define N_IN   256
#define HID    8
#define HEADS  8
#define HC     64     // HEADS*HID
#define OUTC   16
#define NEG_SLOPE 0.2f
#define ENC_NEG_INF 0x007FFFFFu   // enc(-inf)

// ---- order-preserving float<->uint encoding for atomicMax on signed floats ----
__device__ __forceinline__ unsigned enc_f32(float f) {
    unsigned u = __float_as_uint(f);
    return (u & 0x80000000u) ? ~u : (u | 0x80000000u);
}
__device__ __forceinline__ float dec_f32(unsigned e) {
    unsigned u = (e & 0x80000000u) ? (e & 0x7FFFFFFFu) : ~e;
    return __uint_as_float(u);
}

// =============================== init kernels ===============================
__global__ void init1_k(unsigned* __restrict__ mx1, float* __restrict__ den1,
                        float* __restrict__ num1, int n) {
    int i = blockIdx.x * blockDim.x + threadIdx.x;
    if (i < n * HC) num1[i] = 0.f;
    if (i < n * HEADS) { mx1[i] = ENC_NEG_INF; den1[i] = 0.f; }
}

__global__ void init2_k(unsigned* __restrict__ mx2, float* __restrict__ den2,
                        float* __restrict__ num2, int n) {
    int i = blockIdx.x * blockDim.x + threadIdx.x;
    if (i < n * OUTC) num2[i] = 0.f;
    if (i < n) { mx2[i] = ENC_NEG_INF; den2[i] = 0.f; }
}

// =============================== GEMM 1 (WMMA) ==============================
// h1[n,64] = x[n,256] @ W1[256,64]; block = 8 waves, each wave 16 rows x 64 cols
__global__ __launch_bounds__(256) void gemm1_k(const float* __restrict__ x,
                                               const float* __restrict__ W1,
                                               float* __restrict__ h1, int n) {
    __shared__ __align__(32) _Float16 w1t[HC * N_IN];   // [col][k], 32 KB
    int tid = threadIdx.x;
    for (int i = tid; i < HC * N_IN; i += 256) {
        int col = i >> 8, k = i & 255;
        w1t[i] = (_Float16)W1[k * HC + col];
    }
    __syncthreads();

    int wave = tid >> 5, lane = tid & 31;
    int row0  = blockIdx.x * 128 + wave * 16;
    int m     = lane & 15;
    int khalf = lane >> 4;
    int row   = row0 + m;
    int rrow  = (row < n) ? row : (n - 1);
    const float* xr = x + (size_t)rrow * N_IN;

    v8f acc0 = {}, acc1 = {}, acc2 = {}, acc3 = {};
    #pragma unroll
    for (int k0 = 0; k0 < N_IN; k0 += 32) {
        // A fragment: 16x32 f16, k = 16*khalf + s (contiguous, same order as B)
        v16h a;
        int ka = k0 + khalf * 16;       // K 0..7 (lanes 0-15) / 16..23 (lanes 16-31)
        int kb = k0 + 8 + khalf * 16;   // K 8..15 / 24..31
        #pragma unroll
        for (int j = 0; j < 8; ++j) {
            a[j]     = (_Float16)xr[ka + j];
            a[8 + j] = (_Float16)xr[kb + j];
        }
        // B fragments: lane = column, 16 K-contiguous f16 from LDS (32B aligned)
        int koff = k0 + khalf * 16;
        v16h bf0 = *(const v16h*)&w1t[(m +  0) * N_IN + koff];
        v16h bf1 = *(const v16h*)&w1t[(m + 16) * N_IN + koff];
        v16h bf2 = *(const v16h*)&w1t[(m + 32) * N_IN + koff];
        v16h bf3 = *(const v16h*)&w1t[(m + 48) * N_IN + koff];
        acc0 = __builtin_amdgcn_wmma_f32_16x16x32_f16(false, a, false, bf0, (short)0, acc0, false, false);
        acc1 = __builtin_amdgcn_wmma_f32_16x16x32_f16(false, a, false, bf1, (short)0, acc1, false, false);
        acc2 = __builtin_amdgcn_wmma_f32_16x16x32_f16(false, a, false, bf2, (short)0, acc2, false, false);
        acc3 = __builtin_amdgcn_wmma_f32_16x16x32_f16(false, a, false, bf3, (short)0, acc3, false, false);
    }
    #pragma unroll
    for (int r = 0; r < 8; ++r) {
        int orow = row0 + r + 8 * khalf;
        if (orow < n) {
            float* hr = h1 + (size_t)orow * HC;
            hr[m]      = acc0[r];
            hr[m + 16] = acc1[r];
            hr[m + 32] = acc2[r];
            hr[m + 48] = acc3[r];
        }
    }
}

// ===================== attention logit dot-products (layer 1) =====================
__global__ void alpha1_k(const float* __restrict__ h1, const float* __restrict__ a_src,
                         const float* __restrict__ a_dst, float* __restrict__ asrc,
                         float* __restrict__ adst, int n) {
    int i = blockIdx.x * blockDim.x + threadIdx.x;
    if (i >= n * HEADS) return;
    int node = i / HEADS, h = i % HEADS;
    const float* hp = h1 + (size_t)node * HC + h * HID;
    float s = 0.f, d = 0.f;
    #pragma unroll
    for (int c = 0; c < HID; ++c) {
        s += hp[c] * a_src[h * HID + c];
        d += hp[c] * a_dst[h * HID + c];
    }
    asrc[i] = s; adst[i] = d;
}

// =========================== edge segment-max (layer 1) ===========================
__global__ void edge_max1_k(const int* __restrict__ ei, int E, int n,
                            const float* __restrict__ asrc, const float* __restrict__ adst,
                            unsigned* __restrict__ mx) {
    int e = blockIdx.x * blockDim.x + threadIdx.x;
    if (e >= E + n) return;
    int s, d;
    if (e < E) { s = ei[e]; d = ei[E + e]; } else { s = d = e - E; }
    #pragma unroll
    for (int h = 0; h < HEADS; ++h) {
        float v = asrc[s * HEADS + h] + adst[d * HEADS + h];
        v = (v > 0.f) ? v : NEG_SLOPE * v;
        atomicMax(&mx[d * HEADS + h], enc_f32(v));
    }
}

// ============== fused exp / denominator / numerator scatter (layer 1) ==============
__global__ void edge_acc1_k(const int* __restrict__ ei, int E, int n,
                            const float* __restrict__ asrc, const float* __restrict__ adst,
                            const unsigned* __restrict__ mx, const float* __restrict__ h1,
                            float* __restrict__ den, float* __restrict__ num) {
    int i = blockIdx.x * blockDim.x + threadIdx.x;   // (edge, head)
    if (i >= (E + n) * HEADS) return;
    int e = i >> 3, h = i & 7;
    int s, d;
    if (e < E) { s = ei[e]; d = ei[E + e]; } else { s = d = e - E; }
    float v = asrc[s * HEADS + h] + adst[d * HEADS + h];
    v = (v > 0.f) ? v : NEG_SLOPE * v;
    float ex = __expf(v - dec_f32(mx[d * HEADS + h]));
    atomicAdd(&den[d * HEADS + h], ex);
    const float* hp = h1 + (size_t)s * HC + h * HID;
    float*       np = num + (size_t)d * HC + h * HID;
    #pragma unroll
    for (int c = 0; c < HID; ++c) atomicAdd(&np[c], ex * hp[c]);
}

// ===================== normalize + bias + ELU (layer 1 output) =====================
__global__ void finalize1_k(const float* __restrict__ num, const float* __restrict__ den,
                            const float* __restrict__ b1, float* __restrict__ act, int n) {
    int i = blockIdx.x * blockDim.x + threadIdx.x;
    if (i >= n * HC) return;
    int node = i / HC, hc = i % HC, h = hc / HID;
    float v = num[i] / (den[node * HEADS + h] + 1e-16f) + b1[hc];
    act[i] = (v > 0.f) ? v : (__expf(v) - 1.0f);
}

// =============================== GEMM 2 (WMMA) ==============================
// h2[n,16] = act[n,64] @ W2[64,16]
__global__ __launch_bounds__(256) void gemm2_k(const float* __restrict__ act,
                                               const float* __restrict__ W2,
                                               float* __restrict__ h2, int n) {
    __shared__ __align__(32) _Float16 w2t[OUTC * HC];  // [col][k], 2 KB
    int tid = threadIdx.x;
    for (int i = tid; i < OUTC * HC; i += 256) {
        int col = i >> 6, k = i & 63;
        w2t[i] = (_Float16)W2[k * OUTC + col];
    }
    __syncthreads();

    int wave = tid >> 5, lane = tid & 31;
    int row0  = blockIdx.x * 128 + wave * 16;
    int m     = lane & 15;
    int khalf = lane >> 4;
    int row   = row0 + m;
    int rrow  = (row < n) ? row : (n - 1);
    const float* ar = act + (size_t)rrow * HC;

    v8f acc = {};
    #pragma unroll
    for (int k0 = 0; k0 < HC; k0 += 32) {
        v16h a;
        int ka = k0 + khalf * 16, kb = k0 + 8 + khalf * 16;
        #pragma unroll
        for (int j = 0; j < 8; ++j) {
            a[j]     = (_Float16)ar[ka + j];
            a[8 + j] = (_Float16)ar[kb + j];
        }
        int koff = k0 + khalf * 16;
        v16h bf = *(const v16h*)&w2t[m * HC + koff];
        acc = __builtin_amdgcn_wmma_f32_16x16x32_f16(false, a, false, bf, (short)0, acc, false, false);
    }
    #pragma unroll
    for (int r = 0; r < 8; ++r) {
        int orow = row0 + r + 8 * khalf;
        if (orow < n) h2[(size_t)orow * OUTC + m] = acc[r];
    }
}

// ===================== attention dots (layer 2, single head) =====================
__global__ void alpha2_k(const float* __restrict__ h2, const float* __restrict__ a_src2,
                         const float* __restrict__ a_dst2, float* __restrict__ asrc,
                         float* __restrict__ adst, int n) {
    int i = blockIdx.x * blockDim.x + threadIdx.x;
    if (i >= n) return;
    const float* hp = h2 + (size_t)i * OUTC;
    float s = 0.f, d = 0.f;
    #pragma unroll
    for (int c = 0; c < OUTC; ++c) { s += hp[c] * a_src2[c]; d += hp[c] * a_dst2[c]; }
    asrc[i] = s; adst[i] = d;
}

__global__ void edge_max2_k(const int* __restrict__ ei, int E, int n,
                            const float* __restrict__ asrc, const float* __restrict__ adst,
                            unsigned* __restrict__ mx) {
    int e = blockIdx.x * blockDim.x + threadIdx.x;
    if (e >= E + n) return;
    int s, d;
    if (e < E) { s = ei[e]; d = ei[E + e]; } else { s = d = e - E; }
    float v = asrc[s] + adst[d];
    v = (v > 0.f) ? v : NEG_SLOPE * v;
    atomicMax(&mx[d], enc_f32(v));
}

__global__ void edge_acc2_k(const int* __restrict__ ei, int E, int n,
                            const float* __restrict__ asrc, const float* __restrict__ adst,
                            const unsigned* __restrict__ mx, const float* __restrict__ h2,
                            float* __restrict__ den, float* __restrict__ num) {
    int e = blockIdx.x * blockDim.x + threadIdx.x;
    if (e >= E + n) return;
    int s, d;
    if (e < E) { s = ei[e]; d = ei[E + e]; } else { s = d = e - E; }
    float v = asrc[s] + adst[d];
    v = (v > 0.f) ? v : NEG_SLOPE * v;
    float ex = __expf(v - dec_f32(mx[d]));
    atomicAdd(&den[d], ex);
    const float* hp = h2 + (size_t)s * OUTC;
    float*       np = num + (size_t)d * OUTC;
    #pragma unroll
    for (int c = 0; c < OUTC; ++c) atomicAdd(&np[c], ex * hp[c]);
}

// ================== normalize + bias + log_softmax (final output) ==================
__global__ void final_k(const float* __restrict__ num, const float* __restrict__ den,
                        const float* __restrict__ b2, float* __restrict__ out, int n) {
    int i = blockIdx.x * blockDim.x + threadIdx.x;
    if (i >= n) return;
    float inv = 1.0f / (den[i] + 1e-16f);
    float v[OUTC];
    float mx = -3.402823466e+38f;
    #pragma unroll
    for (int c = 0; c < OUTC; ++c) {
        v[c] = num[(size_t)i * OUTC + c] * inv + b2[c];
        mx = fmaxf(mx, v[c]);
    }
    float se = 0.f;
    #pragma unroll
    for (int c = 0; c < OUTC; ++c) se += __expf(v[c] - mx);
    float ls = __logf(se);
    #pragma unroll
    for (int c = 0; c < OUTC; ++c) out[(size_t)i * OUTC + c] = v[c] - mx - ls;
}

// ==================================== launch ====================================
extern "C" void kernel_launch(void* const* d_in, const int* in_sizes, int n_in,
                              void* d_out, int out_size, void* d_ws, size_t ws_size,
                              hipStream_t stream) {
    const float* x      = (const float*)d_in[0];
    const int*   ei     = (const int*)d_in[1];
    const float* W1     = (const float*)d_in[2];
    const float* a_src1 = (const float*)d_in[3];
    const float* a_dst1 = (const float*)d_in[4];
    const float* b1     = (const float*)d_in[5];
    const float* W2     = (const float*)d_in[6];
    const float* a_src2 = (const float*)d_in[7];
    const float* a_dst2 = (const float*)d_in[8];
    const float* b2     = (const float*)d_in[9];

    int n = in_sizes[0] / N_IN;
    int E = in_sizes[1] / 2;
    int ET = E + n;

    // workspace layout (floats): 160 per node total (layer-2 aliases dead num1)
    float*    ws    = (float*)d_ws;
    float*    h1    = ws;                                   // n*64 (later reused as act1)
    float*    asrc1 = h1 + (size_t)n * HC;                  // n*8
    float*    adst1 = asrc1 + (size_t)n * HEADS;            // n*8
    unsigned* mx1   = (unsigned*)(adst1 + (size_t)n * HEADS); // n*8
    float*    den1  = (float*)(mx1 + (size_t)n * HEADS);    // n*8
    float*    num1  = den1 + (size_t)n * HEADS;             // n*64 (layer-2 region)
    float*    h2    = num1;                                 // n*16
    float*    num2  = num1 + (size_t)n * OUTC;              // n*16
    float*    asrc2 = num1 + (size_t)n * 32;                // n
    float*    adst2 = asrc2 + n;                            // n
    float*    den2  = adst2 + n;                            // n
    unsigned* mx2   = (unsigned*)(den2 + n);                // n

    dim3 blk(256);
    // ---------------- layer 1 ----------------
    init1_k    <<<(n * HC + 255) / 256, blk, 0, stream>>>(mx1, den1, num1, n);
    gemm1_k    <<<(n + 127) / 128,      blk, 0, stream>>>(x, W1, h1, n);
    alpha1_k   <<<(n * HEADS + 255) / 256, blk, 0, stream>>>(h1, a_src1, a_dst1, asrc1, adst1, n);
    edge_max1_k<<<(ET + 255) / 256,     blk, 0, stream>>>(ei, E, n, asrc1, adst1, mx1);
    edge_acc1_k<<<(ET * HEADS + 255) / 256, blk, 0, stream>>>(ei, E, n, asrc1, adst1, mx1, h1, den1, num1);
    finalize1_k<<<(n * HC + 255) / 256, blk, 0, stream>>>(num1, den1, b1, h1, n);  // act1 aliases h1
    // ---------------- layer 2 ----------------
    gemm2_k    <<<(n + 127) / 128,      blk, 0, stream>>>(h1, W2, h2, n);
    alpha2_k   <<<(n + 255) / 256,      blk, 0, stream>>>(h2, a_src2, a_dst2, asrc2, adst2, n);
    init2_k    <<<(n * OUTC + 255) / 256, blk, 0, stream>>>(mx2, den2, num2, n);
    edge_max2_k<<<(ET + 255) / 256,     blk, 0, stream>>>(ei, E, n, asrc2, adst2, mx2);
    edge_acc2_k<<<(ET + 255) / 256,     blk, 0, stream>>>(ei, E, n, asrc2, adst2, mx2, h2, den2, num2);
    final_k    <<<(n + 255) / 256,      blk, 0, stream>>>(num2, den2, b2, (float*)d_out, n);
}
